// BotImpact_15693810499989
// MI455X (gfx1250) — matrix operand order repaired and checked by reading, not compiled, not run on hardware
//
#include <hip/hip_runtime.h>
#include <math.h>

// ---------------- problem constants (from reference) ----------------
#define N_NODES 50000
#define N_EDGES 800000
#define IN_DIM  128
#define H_DIM   64
#define HEADS   2
#define HDIM    128      // HEADS * H_DIM
#define N_IDX   5000
#define M_TILES (N_NODES / 16)   // 3125, exact

typedef float    v8f  __attribute__((ext_vector_type(8)));
typedef _Float16 v8h  __attribute__((ext_vector_type(8)));
typedef _Float16 v16h __attribute__((ext_vector_type(16)));

__device__ __forceinline__ float leaky(float x, float s) { return x > 0.f ? x : s * x; }

__device__ __forceinline__ void atomicMaxF(float* addr, float val) {
    int* ia = (int*)addr;
    int old = *ia;
    while (__int_as_float(old) < val) {
        int assumed = old;
        old = atomicCAS(ia, assumed, __float_as_int(val));
        if (old == assumed) break;
    }
}

// ---------------- utility kernels ----------------
__global__ void fill_f32(float* __restrict__ p, long n, float v) {
    long i = blockIdx.x * (long)blockDim.x + threadIdx.x;
    if (i < n) p[i] = v;
}

__global__ void cvt_f16(const float* __restrict__ s, _Float16* __restrict__ d, long n) {
    long i = blockIdx.x * (long)blockDim.x + threadIdx.x;
    if (i < n) d[i] = (_Float16)s[i];
}

// W: [128,128] row-major (K x C)  ->  Wt: [C][K] f16 (so a lane's K-run is contiguous)
__global__ void cvt_wt(const float* __restrict__ W, _Float16* __restrict__ Wt) {
    int i = blockIdx.x * blockDim.x + threadIdx.x;
    if (i < 128 * 128) {
        int c = i >> 7, k = i & 127;
        Wt[c * 128 + k] = (_Float16)W[k * 128 + c];
    }
}

// ---------------- WMMA GEMM: C[M,128] = Xh[M,128] * W, K=128 ----------------
// one wave per 16x16 tile; 8 waves/block -> 128 output columns; grid.x = M/16
__global__ __launch_bounds__(256) void wmma_gemm_128(
    const _Float16* __restrict__ Xh,   // [M,128] row-major f16
    const _Float16* __restrict__ Wt,   // [128,128] f16, Wt[c*128+k] = W[k][c]
    float* __restrict__ C)             // [M,128] f32
{
    const int K = 128;
    int wave = threadIdx.x >> 5;
    int lane = threadIdx.x & 31;
    int lh   = lane >> 4;      // lane half (0/1)
    int r    = lane & 15;      // row within A tile / column within B,C tile
    long mBase = (long)blockIdx.x * 16;
    int  nBase = wave * 16;

    const _Float16* arow = Xh + (mBase + r) * K;
    const _Float16* bcol = Wt + (long)(nBase + r) * K;

    v8f acc = {};
    #pragma unroll
    for (int ks = 0; ks < K; ks += 32) {
        // A 16x32 f16 layout: lane-half lh holds k = ks+lh*8+[0..7] (e0..7) and +16 (e8..15)
        v8h alo = *(const v8h*)(arow + ks + lh * 8);
        v8h ahi = *(const v8h*)(arow + ks + lh * 8 + 16);
        // B 32x16 f16 layout: lane-half lh holds k = ks+lh*8+[0..7] (e0..7) and +16 (e8..15) for column r
        v8h blo = *(const v8h*)(bcol + ks + lh * 8);
        v8h bhi = *(const v8h*)(bcol + ks + lh * 8 + 16);
        v16h a = __builtin_shufflevector(alo, ahi, 0,1,2,3,4,5,6,7,8,9,10,11,12,13,14,15);
        v16h b = __builtin_shufflevector(blo, bhi, 0,1,2,3,4,5,6,7,8,9,10,11,12,13,14,15);
        acc = __builtin_amdgcn_wmma_f32_16x16x32_f16(
            /*neg_a=*/false, a, /*neg_b=*/false, b,
            /*c_mod=*/(short)0, acc, /*reuse_a=*/false, /*reuse_b=*/false);
    }
    // C/D layout: element j -> row mBase + lh*8 + j, col nBase + r
    float* crow = C + mBase * HDIM + nBase + r;
    #pragma unroll
    for (int j = 0; j < 8; ++j)
        crow[(long)(j + lh * 8) * HDIM] = acc[j];
}

// ---------------- GAT attention kernels ----------------
__global__ void alpha_kernel(const float* __restrict__ H,
                             const float* __restrict__ a_src,
                             const float* __restrict__ a_dst,
                             float* __restrict__ as, float* __restrict__ ad) {
    int i = blockIdx.x * blockDim.x + threadIdx.x;          // over N*HEADS
    if (i < N_NODES * HEADS) {
        int n = i / HEADS, h = i % HEADS;
        const float* hp = H + (long)n * HDIM + h * H_DIM;
        float ss = 0.f, sd = 0.f;
        #pragma unroll 4
        for (int d = 0; d < H_DIM; ++d) {
            float v = hp[d];
            ss += v * a_src[h * H_DIM + d];
            sd += v * a_dst[h * H_DIM + d];
        }
        as[i] = ss; ad[i] = sd;
    }
}

__device__ __forceinline__ void edge_endpoints(const int* ei, long i, int& src, int& dst) {
    if (i < N_EDGES) { src = ei[i]; dst = ei[N_EDGES + i]; }
    else             { src = dst = (int)(i - N_EDGES); }     // self loop
}

__global__ void edge_max_kernel(const int* __restrict__ ei,
                                const float* __restrict__ as, const float* __restrict__ ad,
                                float* __restrict__ m, float* __restrict__ ew) {
    long t = blockIdx.x * (long)blockDim.x + threadIdx.x;   // over (E+N)*HEADS
    long total = (long)(N_EDGES + N_NODES) * HEADS;
    if (t < total) {
        long i = t >> 1; int h = (int)(t & 1);
        int src, dst; edge_endpoints(ei, i, src, dst);
        float e = leaky(as[src * HEADS + h] + ad[dst * HEADS + h], 0.2f);
        ew[t] = e;
        atomicMaxF(&m[dst * HEADS + h], e);
    }
}

__global__ void edge_expsum_kernel(const int* __restrict__ ei,
                                   const float* __restrict__ m,
                                   float* __restrict__ denom, float* __restrict__ ew) {
    long t = blockIdx.x * (long)blockDim.x + threadIdx.x;
    long total = (long)(N_EDGES + N_NODES) * HEADS;
    if (t < total) {
        long i = t >> 1; int h = (int)(t & 1);
        int src, dst; edge_endpoints(ei, i, src, dst);
        (void)src;
        float ex = __expf(ew[t] - m[dst * HEADS + h]);
        ew[t] = ex;
        atomicAdd(&denom[dst * HEADS + h], ex);
    }
}

// one thread per (edge, 4-float chunk); 32 chunks cover HDIM=128
__global__ void edge_agg_kernel(const int* __restrict__ ei,
                                const float* __restrict__ ew, const float* __restrict__ denom,
                                const float* __restrict__ H, float* __restrict__ agg) {
    long t = blockIdx.x * (long)blockDim.x + threadIdx.x;
    long total = (long)(N_EDGES + N_NODES) * 32;
    if (t < total) {
        long i = t >> 5; int p = (int)(t & 31);
        int src, dst; edge_endpoints(ei, i, src, dst);
        int c0 = p * 4;
        int h  = c0 >> 6;                                    // head for this chunk
        float alpha = ew[i * HEADS + h] / denom[dst * HEADS + h];
        const float4 hv = *(const float4*)(H + (long)src * HDIM + c0);
        float* ap = agg + (long)dst * HDIM + c0;
        atomicAdd(ap + 0, alpha * hv.x);
        atomicAdd(ap + 1, alpha * hv.y);
        atomicAdd(ap + 2, alpha * hv.z);
        atomicAdd(ap + 3, alpha * hv.w);
    }
}

__global__ void bias_act_kernel(float* __restrict__ A, const float* __restrict__ b, int do_relu) {
    long i = blockIdx.x * (long)blockDim.x + threadIdx.x;   // over N*HDIM
    if (i < (long)N_NODES * HDIM) {
        int c = (int)(i & 127);
        float v = A[i] + b[c];
        A[i] = do_relu ? fmaxf(v, 0.f) : v;
    }
}

// ---------------- MLP heads ----------------
__global__ void mlp_gather_128x64(const float* __restrict__ X, const int* __restrict__ idx,
                                  const float* __restrict__ W, const float* __restrict__ b,
                                  float* __restrict__ out) {
    int t = blockIdx.x * blockDim.x + threadIdx.x;          // over N_IDX*64
    if (t < N_IDX * H_DIM) {
        int i = t / H_DIM, j = t % H_DIM;
        const float* xr = X + (long)idx[i] * HDIM;
        float s = b[j];
        #pragma unroll 4
        for (int k = 0; k < HDIM; ++k) s += xr[k] * W[k * H_DIM + j];
        out[t] = leaky(s, 0.01f);
    }
}

__global__ void mlp_64x1(const float* __restrict__ hid, const float* __restrict__ W,
                         const float* __restrict__ b, float* __restrict__ out) {
    int i = blockIdx.x * blockDim.x + threadIdx.x;
    if (i < N_IDX) {
        float s = b[0];
        #pragma unroll 4
        for (int k = 0; k < H_DIM; ++k) s += hid[i * H_DIM + k] * W[k];
        out[i] = leaky(s, 0.01f);
    }
}

__global__ void tprob_kernel(const float* __restrict__ X, const float* __restrict__ Wp,
                             const float* __restrict__ bp, float* __restrict__ out) {
    int t = blockIdx.x * blockDim.x + threadIdx.x;          // over N*2
    if (t < N_NODES * 2) {
        int n = t >> 1, j = t & 1;
        const float* xr = X + (long)n * HDIM;
        float s = bp[j];
        #pragma unroll 4
        for (int k = 0; k < HDIM; ++k) s += xr[k] * Wp[k * 2 + j];
        out[t] = leaky(s, 0.01f);
    }
}

// ---------------- host-side orchestration ----------------
static inline int blocks_for(long n, int bs) { return (int)((n + bs - 1) / bs); }

struct Scratch {
    _Float16* Xh;  _Float16* Wt;
    float *H, *Z1, *as, *ad, *m, *denom, *ew, *hid;
};

static void run_gat_layer(const float* Xin, const int* ei,
                          const float* W, const float* a_src, const float* a_dst,
                          const float* bias, float* outAgg, int do_relu,
                          const Scratch& S, hipStream_t stream) {
    const long NX = (long)N_NODES * 128;
    const long NEh = (long)(N_EDGES + N_NODES) * HEADS;
    cvt_f16<<<blocks_for(NX, 256), 256, 0, stream>>>(Xin, S.Xh, NX);
    cvt_wt<<<blocks_for(128 * 128, 256), 256, 0, stream>>>(W, S.Wt);
    wmma_gemm_128<<<M_TILES, 256, 0, stream>>>(S.Xh, S.Wt, S.H);
    alpha_kernel<<<blocks_for((long)N_NODES * HEADS, 256), 256, 0, stream>>>(S.H, a_src, a_dst, S.as, S.ad);
    fill_f32<<<blocks_for((long)N_NODES * HEADS, 256), 256, 0, stream>>>(S.m, (long)N_NODES * HEADS, -1e30f);
    fill_f32<<<blocks_for((long)N_NODES * HEADS, 256), 256, 0, stream>>>(S.denom, (long)N_NODES * HEADS, 0.f);
    fill_f32<<<blocks_for(NX, 256), 256, 0, stream>>>(outAgg, NX, 0.f);
    edge_max_kernel<<<blocks_for(NEh, 256), 256, 0, stream>>>(ei, S.as, S.ad, S.m, S.ew);
    edge_expsum_kernel<<<blocks_for(NEh, 256), 256, 0, stream>>>(ei, S.m, S.denom, S.ew);
    edge_agg_kernel<<<blocks_for((long)(N_EDGES + N_NODES) * 32, 256), 256, 0, stream>>>(
        ei, S.ew, S.denom, S.H, outAgg);
    bias_act_kernel<<<blocks_for(NX, 256), 256, 0, stream>>>(outAgg, bias, do_relu);
}

extern "C" void kernel_launch(void* const* d_in, const int* in_sizes, int n_in,
                              void* d_out, int out_size, void* d_ws, size_t ws_size,
                              hipStream_t stream) {
    // inputs (setup_inputs order)
    const float* x        = (const float*)d_in[0];
    const int*   ei       = (const int*)  d_in[1];
    const float* fx       = (const float*)d_in[2];
    const int*   fei      = (const int*)  d_in[3];
    const int*   treat    = (const int*)  d_in[4];
    const int*   control  = (const int*)  d_in[5];
    const float* W1       = (const float*)d_in[6];
    const float* a_src1   = (const float*)d_in[7];
    const float* a_dst1   = (const float*)d_in[8];
    const float* b1       = (const float*)d_in[9];
    const float* W2       = (const float*)d_in[10];
    const float* a_src2   = (const float*)d_in[11];
    const float* a_dst2   = (const float*)d_in[12];
    const float* b2       = (const float*)d_in[13];
    const float* WyS      = (const float*)d_in[14];
    const float* byS      = (const float*)d_in[15];
    const float* Wy1      = (const float*)d_in[16];
    const float* by1      = (const float*)d_in[17];
    const float* Wy0      = (const float*)d_in[18];
    const float* by0      = (const float*)d_in[19];
    const float* Wp       = (const float*)d_in[20];
    const float* bp       = (const float*)d_in[21];

    // outputs: y1(5000), yc0(5000), y0(5000), yc1(5000), xZ2(6.4M), xfZ2(6.4M), tprob(100000)
    float* out   = (float*)d_out;
    float* y1    = out;
    float* yc0   = out + N_IDX;
    float* y0    = out + 2 * N_IDX;
    float* yc1   = out + 3 * N_IDX;
    float* xZ2   = out + 4 * N_IDX;
    float* xfZ2  = xZ2 + (long)N_NODES * HDIM;
    float* tp    = xfZ2 + (long)N_NODES * HDIM;

    // bump-allocate scratch (all 256B aligned)
    char* base = (char*)d_ws;
    size_t off = 0;
    auto alloc = [&](size_t bytes) -> void* {
        void* p = base + off;
        off = (off + bytes + 255) & ~(size_t)255;
        return p;
    };
    Scratch S;
    S.Xh    = (_Float16*)alloc((size_t)N_NODES * 128 * sizeof(_Float16));
    S.Wt    = (_Float16*)alloc((size_t)128 * 128 * sizeof(_Float16));
    S.H     = (float*)   alloc((size_t)N_NODES * 128 * sizeof(float));
    S.Z1    = (float*)   alloc((size_t)N_NODES * 128 * sizeof(float));
    S.as    = (float*)   alloc((size_t)N_NODES * HEADS * sizeof(float));
    S.ad    = (float*)   alloc((size_t)N_NODES * HEADS * sizeof(float));
    S.m     = (float*)   alloc((size_t)N_NODES * HEADS * sizeof(float));
    S.denom = (float*)   alloc((size_t)N_NODES * HEADS * sizeof(float));
    S.ew    = (float*)   alloc((size_t)(N_EDGES + N_NODES) * HEADS * sizeof(float));
    S.hid   = (float*)   alloc((size_t)N_IDX * H_DIM * sizeof(float));
    (void)ws_size; (void)in_sizes; (void)n_in; (void)out_size;

    // real graph: layer1 -> Z1 (ReLU), layer2 -> xZ2
    run_gat_layer(x,    ei,  W1, a_src1, a_dst1, b1, S.Z1, 1, S, stream);
    run_gat_layer(S.Z1, ei,  W2, a_src2, a_dst2, b2, xZ2,  0, S, stream);
    // fake graph: layer1 -> Z1 (reuse), layer2 -> xfZ2
    run_gat_layer(fx,   fei, W1, a_src1, a_dst1, b1, S.Z1, 1, S, stream);
    run_gat_layer(S.Z1, fei, W2, a_src2, a_dst2, b2, xfZ2, 0, S, stream);

    // heads (sequential reuse of S.hid on same stream)
    const int hb = blocks_for((long)N_IDX * H_DIM, 256);
    const int ob = blocks_for((long)N_IDX, 256);
    mlp_gather_128x64<<<hb, 256, 0, stream>>>(xZ2,  treat,   WyS, byS, S.hid);
    mlp_64x1<<<ob, 256, 0, stream>>>(S.hid, Wy1, by1, y1);
    mlp_gather_128x64<<<hb, 256, 0, stream>>>(xfZ2, treat,   WyS, byS, S.hid);
    mlp_64x1<<<ob, 256, 0, stream>>>(S.hid, Wy0, by0, yc0);
    mlp_gather_128x64<<<hb, 256, 0, stream>>>(xZ2,  control, WyS, byS, S.hid);
    mlp_64x1<<<ob, 256, 0, stream>>>(S.hid, Wy0, by0, y0);
    mlp_gather_128x64<<<hb, 256, 0, stream>>>(xfZ2, control, WyS, byS, S.hid);
    mlp_64x1<<<ob, 256, 0, stream>>>(S.hid, Wy1, by1, yc1);

    tprob_kernel<<<blocks_for((long)N_NODES * 2, 256), 256, 0, stream>>>(xZ2, Wp, bp, tp);
}
